// BiLSTM_9079560864425
// MI455X (gfx1250) — matrix operand order repaired and checked by reading, not compiled, not run on hardware
//
#include <hip/hip_runtime.h>
#include <hip/hip_bf16.h>

// ---------------------------------------------------------------------------
// BiLSTM (L=3, H=512, D_IN=1024, B=64, T=512) for MI455X (gfx1250, wave32).
// bf16 V_WMMA_F32_16X16X32_BF16 GEMMs; weights bf16 + L2-resident; persistent
// per-layer kernel; c/h in registers; per-step grid barrier. Activations are
// staged per-block into LDS via async global->LDS copies (ASYNCcnt); weight
// (B) fragments stream from L2, double-buffered one chunk deep. One 16x16
// batch tile per wave (~160 VGPRs) keeps the allocator off the 256 ceiling.
// ---------------------------------------------------------------------------

typedef __attribute__((ext_vector_type(8)))  float          v8f;
typedef __attribute__((ext_vector_type(16))) __bf16         v16bf;
typedef __attribute__((ext_vector_type(8)))  unsigned short v8u;
typedef __attribute__((ext_vector_type(16))) unsigned short v16u;
typedef __attribute__((ext_vector_type(4)))  int            v4i;

#define T_   512
#define B_   64
#define H_   512
#define DIN_ 1024
#define L_   3
#define NBLK 32
#define NTHR 256   // 8 waves/block, 256 waves total

#define SX_BYTES (B_ * DIN_ * 2)        // 131072: x_t tile, 64 x 1024 bf16
#define SH_BYTES (B_ * H_ * 2)          //  65536: h tile,   64 x 512  bf16
#define SMEM_BYTES (SX_BYTES + SH_BYTES)

#if __has_builtin(__builtin_amdgcn_global_load_async_to_lds_b128) && \
    __has_builtin(__builtin_amdgcn_s_wait_asynccnt)
#define USE_ASYNC_LDS 1
#else
#define USE_ASYNC_LDS 0
#endif

#define GLOBAL_AS __attribute__((address_space(1)))
#define LDS_AS    __attribute__((address_space(3)))

__device__ __forceinline__ unsigned short f2bf(float f) {
  unsigned u = __builtin_bit_cast(unsigned, f);
  unsigned r = u + 0x7fffu + ((u >> 16) & 1u);   // round-to-nearest-even
  return (unsigned short)(r >> 16);
}

__device__ __forceinline__ float sigm(float x) { return 1.0f / (1.0f + __expf(-x)); }

__device__ __forceinline__ v8f wmma_bf16(v16bf a, v16bf b, v8f c) {
  // D(f32 16x16) = A(16x32 bf16) * B(32x16 bf16) + C
  return __builtin_amdgcn_wmma_f32_16x16x32_bf16(false, a, false, b, (short)0, c,
                                                 false, false);
}

// A fragment: 16x32 bf16 tile, lane&15 = row M, lane>>4 selects K-halves.
// Per-lane data = row[k0+koff .. +15] (contiguous), koff=(lane>>4)*16.
__device__ __forceinline__ v16bf a_frag(const unsigned short* m, int ld, int row,
                                        int k0, int lane) {
  const int koff = (lane >> 4) * 16;
  const unsigned short* p = m + (long)row * ld + k0 + koff;
  v8u lo = *(const v8u*)p;
  v8u hi = *(const v8u*)(p + 8);
  v16u c = __builtin_shufflevector(lo, hi, 0, 1, 2, 3, 4, 5, 6, 7,
                                           8, 9, 10, 11, 12, 13, 14, 15);
  return __builtin_bit_cast(v16bf, c);
}

// B fragment: 32x16 bf16 tile of W^T. W stored row-major (N,K), so column n of
// B is row n of W. lane&15 = column N, lane>>4 selects K-half (16 contiguous).
__device__ __forceinline__ v16bf b_frag(const unsigned short* __restrict__ w,
                                        int ld, int nrow, int k0, int lane) {
  const unsigned short* p = w + (long)nrow * ld + k0 + (lane >> 4) * 16;
  v16u c = *(const v16u*)p;
  return __builtin_bit_cast(v16bf, c);
}

// Weight fragments for one 32-wide K chunk: B tiles for all four gates.
struct BFrags {
  v16bf b[4];
};

__device__ __forceinline__ BFrags load_b(const unsigned short* __restrict__ W,
                                         int ldw, int nbase, int k0, int lane) {
  BFrags f;
#pragma unroll
  for (int g = 0; g < 4; ++g) f.b[g] = b_frag(W, ldw, g * H_ + nbase, k0, lane);
  return f;
}

// A from LDS (ds_load) + 4 WMMAs for one K chunk (one 16-row batch tile).
__device__ __forceinline__ void do_chunk(const unsigned short* sA, int ldA,
                                         int mbase, int k0, int lane,
                                         const BFrags& f, v8f (&acc)[4]) {
  v16bf a = a_frag(sA, ldA, mbase + (lane & 15), k0, lane);
#pragma unroll
  for (int g = 0; g < 4; ++g) acc[g] = wmma_bf16(a, f.b[g], acc[g]);
}

// -------------------------- conversion kernel ------------------------------
__global__ void f32_to_bf16_kernel(const float* __restrict__ src,
                                   unsigned short* __restrict__ dst, long n) {
  long i = (long)blockIdx.x * blockDim.x + threadIdx.x;
  long stride = (long)gridDim.x * blockDim.x;
  for (; i < n; i += stride) dst[i] = f2bf(src[i]);
}

// -------------------------- grid-wide barrier ------------------------------
__device__ __forceinline__ void grid_barrier(unsigned* __restrict__ bar,
                                             unsigned target) {
  __syncthreads();
  if (threadIdx.x == 0) {
    __threadfence();  // release: make this block's h stores visible device-wide
    __hip_atomic_fetch_add(bar, 1u, __ATOMIC_RELEASE, __HIP_MEMORY_SCOPE_AGENT);
    while (__hip_atomic_load(bar, __ATOMIC_ACQUIRE, __HIP_MEMORY_SCOPE_AGENT) <
           target) {
      __builtin_amdgcn_s_sleep(1);
    }
  }
  __syncthreads();
  __threadfence();    // acquire: invalidate stale near-cache lines for h reads
}

// ---------------- per-step LDS staging of x_t and h_{t-1} ------------------
__device__ __forceinline__ void stage_to_lds(unsigned short* sX,
                                             unsigned short* sH,
                                             const unsigned short* Xt,
                                             const unsigned short* hrd) {
  const int tid = threadIdx.x;
#if USE_ASYNC_LDS
  const GLOBAL_AS char* gx = (const GLOBAL_AS char*)Xt;
  const GLOBAL_AS char* gh = (const GLOBAL_AS char*)hrd;
  LDS_AS char* lx = (LDS_AS char*)sX;
  LDS_AS char* lh = (LDS_AS char*)sH;
#pragma unroll
  for (int i = 0; i < SX_BYTES / 16 / NTHR; ++i) {      // 32 iters
    int off = (tid + i * NTHR) * 16;
    __builtin_amdgcn_global_load_async_to_lds_b128((GLOBAL_AS v4i*)(gx + off),
                                                   (LDS_AS v4i*)(lx + off), 0, 0);
  }
#pragma unroll
  for (int i = 0; i < SH_BYTES / 16 / NTHR; ++i) {      // 16 iters
    int off = (tid + i * NTHR) * 16;
    __builtin_amdgcn_global_load_async_to_lds_b128((GLOBAL_AS v4i*)(gh + off),
                                                   (LDS_AS v4i*)(lh + off), 0, 0);
  }
  __builtin_amdgcn_s_wait_asynccnt(0);
#else
  for (int i = tid; i < SX_BYTES / 16; i += NTHR)
    ((v16u*)sX)[i] = ((const v16u*)Xt)[i];
  for (int i = tid; i < SH_BYTES / 16; i += NTHR)
    ((v16u*)sH)[i] = ((const v16u*)hrd)[i];
#endif
  __syncthreads();
}

// -------------------------- persistent LSTM layer --------------------------
// One launch runs both directions of one layer over all T steps.
// Wave task: dir (2) x h-column tile (32 of 16) x batch tile (4 of 16 rows).
// dir is wid bit 7, so it is uniform within each 8-wave block.
__global__ void __launch_bounds__(NTHR)
lstm_layer_kernel(const unsigned short* __restrict__ Xin,  // (T,B,1024) bf16
                  unsigned short* __restrict__ Xout,       // (T,B,1024) bf16 | null
                  float* __restrict__ OutX,                // (T,B,1024) f32 | null
                  const unsigned short* __restrict__ WihL, // (2,4H,DIN) bf16
                  const unsigned short* __restrict__ WhhL, // (2,4H,H) bf16
                  const float* __restrict__ bihL,          // (2,4H) f32
                  const float* __restrict__ bhhL,          // (2,4H) f32
                  const int* __restrict__ lengths,         // (B)
                  unsigned short* __restrict__ hbuf,       // (2 dir,2 parity,B,H)
                  float* __restrict__ hn_out,              // (6,B,H)
                  float* __restrict__ cn_out,              // (6,B,H)
                  unsigned* __restrict__ bar, int layer) {
  extern __shared__ unsigned short smem[];
  unsigned short* sX = smem;                      // 64 x 1024 bf16
  unsigned short* sH = smem + SX_BYTES / 2;       // 64 x 512  bf16
  __shared__ int slen[B_];
  if (threadIdx.x < B_) slen[threadIdx.x] = lengths[threadIdx.x];
  __syncthreads();

  const int lane  = threadIdx.x & 31;
  const int wid   = (blockIdx.x * blockDim.x + threadIdx.x) >> 5;  // 0..255
  const int mt    = wid & 3;          // batch rows [mt*16, mt*16+16)
  const int ncol  = (wid >> 2) & 31;  // h-column tile
  const int dir   = (wid >> 7) & 1;   // 0 fwd, 1 bwd (uniform within a block)

  const int mbase = mt * 16;
  const int hc0   = ncol * 16;
  const int col   = lane & 15;        // C-layout column within tile
  const int rhalf = (lane >> 4) * 8;  // C-layout row offset for lanes 16..31
  const int hcol  = hc0 + col;
  const int nb    = hc0 + col;        // per-lane B-matrix row base

  const unsigned short* Wih = WihL + (long)dir * 4 * H_ * DIN_;
  const unsigned short* Whh = WhhL + (long)dir * 4 * H_ * H_;
  unsigned short* hb0 = hbuf + ((long)dir * 2 + 0) * (B_ * H_);
  unsigned short* hb1 = hbuf + ((long)dir * 2 + 1) * (B_ * H_);

  // Bias per gate: in C/D layout each lane's 8 elements share one column n.
  float bias[4];
#pragma unroll
  for (int g = 0; g < 4; ++g) {
    int n = g * H_ + hcol;
    bias[g] = bihL[dir * 4 * H_ + n] + bhhL[dir * 4 * H_ + n];
  }

  // Persistent cell/hidden state (masked-carry semantics), one 16x16 tile.
  float c[8], h[8];
#pragma unroll
  for (int r = 0; r < 8; ++r) { c[r] = 0.0f; h[r] = 0.0f; }

  // Zero-init both h parities for this wave's tile (h0 = 0).
#pragma unroll
  for (int r = 0; r < 8; ++r) {
    int brow = mbase + r + rhalf;
    hb0[brow * H_ + hcol] = 0;
    hb1[brow * H_ + hcol] = 0;
  }

  unsigned nbar = 0;
  grid_barrier(bar, ++nbar * gridDim.x);   // h init visible everywhere

  for (int s = 0; s < T_; ++s) {
    const int t = dir ? (T_ - 1 - s) : s;  // backward walks time in reverse
    const unsigned short* Xt  = Xin + (long)t * B_ * DIN_;
    const unsigned short* hrd = (s & 1) ? hb1 : hb0;
    unsigned short*       hwr = (s & 1) ? hb0 : hb1;

    // Cooperative async stage of this step's activations into LDS.
    stage_to_lds(sX, sH, Xt, hrd);

    // Hint next timestep's activation rows toward L2 while we compute.
    if (s + 1 < T_) {
      const int tn = dir ? (T_ - 2 - s) : (s + 1);
      __builtin_prefetch(Xin + ((long)tn * B_ + (threadIdx.x & 63)) * DIN_, 0, 1);
    }

    v8f acc[4];
#pragma unroll
    for (int g = 0; g < 4; ++g)
#pragma unroll
      for (int r = 0; r < 8; ++r) acc[g][r] = bias[g];

    // ---- K loop: A from LDS, B weights double-buffered from L2 ----
    BFrags bcur = load_b(Wih, DIN_, nb, 0, lane);

    for (int k0 = 0; k0 < DIN_ - 32; k0 += 32) {   // X phase (31 full iters)
      BFrags bnxt = load_b(Wih, DIN_, nb, k0 + 32, lane);
      do_chunk(sX, DIN_, mbase, k0, lane, bcur, acc);
      bcur = bnxt;
    }
    {  // phase handoff: overlap last X chunk with first Whh loads
      BFrags bnxt = load_b(Whh, H_, nb, 0, lane);
      do_chunk(sX, DIN_, mbase, DIN_ - 32, lane, bcur, acc);
      bcur = bnxt;
    }
    for (int k0 = 0; k0 < H_ - 32; k0 += 32) {     // H phase (15 full iters)
      BFrags bnxt = load_b(Whh, H_, nb, k0 + 32, lane);
      do_chunk(sH, H_, mbase, k0, lane, bcur, acc);
      bcur = bnxt;
    }
    do_chunk(sH, H_, mbase, H_ - 32, lane, bcur, acc);   // drain

    // Nonlinearities + cell update in registers (C-matrix lane layout).
#pragma unroll
    for (int r = 0; r < 8; ++r) {
      int brow = mbase + r + rhalf;
      float iv = sigm(acc[0][r]);
      float fv = sigm(acc[1][r]);
      float gv = tanhf(acc[2][r]);
      float ov = sigm(acc[3][r]);
      float cn = fv * c[r] + iv * gv;
      float hn = ov * tanhf(cn);
      bool act = (t < slen[brow]);
      c[r] = act ? cn : c[r];
      h[r] = act ? hn : h[r];
      float outv = act ? hn : 0.0f;
      hwr[brow * H_ + hcol] = f2bf(h[r]);
      long oidx = ((long)t * B_ + brow) * (2 * H_) + (long)dir * H_ + hcol;
      if (Xout) Xout[oidx] = f2bf(outv);  // bf16 input for next layer
      if (OutX) OutX[oidx] = outv;        // final-layer fp32 output
    }

    grid_barrier(bar, ++nbar * gridDim.x);
  }

  // Final h_n / c_n straight from registers.
#pragma unroll
  for (int r = 0; r < 8; ++r) {
    int brow = mbase + r + rhalf;
    long sidx = ((long)(2 * layer + dir) * B_ + brow) * H_ + hcol;
    hn_out[sidx] = h[r];
    cn_out[sidx] = c[r];
  }
}

// ------------------------------ host side ----------------------------------
extern "C" void kernel_launch(void* const* d_in, const int* in_sizes, int n_in,
                              void* d_out, int out_size, void* d_ws,
                              size_t ws_size, hipStream_t stream) {
  (void)in_sizes; (void)n_in; (void)out_size; (void)ws_size;
  const float* x   = (const float*)d_in[0];
  const int*   len = (const int*)d_in[1];
  const float* Wih = (const float*)d_in[2];
  const float* Whh = (const float*)d_in[3];
  const float* bih = (const float*)d_in[4];
  const float* bhh = (const float*)d_in[5];
  float* out = (float*)d_out;

  const long WIH_E  = (long)L_ * 2 * 4 * H_ * DIN_;  // 12,582,912
  const long WHH_E  = (long)L_ * 2 * 4 * H_ * H_;    //  6,291,456
  const long X_E    = (long)T_ * B_ * DIN_;          // 33,554,432
  const long HBUF_E = 2L * 2 * B_ * H_;              //    131,072

  char* ws = (char*)d_ws;
  size_t off = 0;
  auto carve = [&](size_t bytes) -> void* {
    void* p = ws + off;
    off = (off + bytes + 255) & ~(size_t)255;
    return p;
  };
  unsigned*       bar   = (unsigned*)carve(256);
  unsigned short* WihB  = (unsigned short*)carve(WIH_E * 2);
  unsigned short* WhhB  = (unsigned short*)carve(WHH_E * 2);
  unsigned short* X0    = (unsigned short*)carve(X_E * 2);
  unsigned short* X1    = (unsigned short*)carve(X_E * 2);
  unsigned short* hbuf  = (unsigned short*)carve(HBUF_E * 2);

  // One-time-per-launch bf16 conversion: weights + layer-0 input (L2-resident).
  f32_to_bf16_kernel<<<2048, 256, 0, stream>>>(Wih, WihB, WIH_E);
  f32_to_bf16_kernel<<<2048, 256, 0, stream>>>(Whh, WhhB, WHH_E);
  f32_to_bf16_kernel<<<4096, 256, 0, stream>>>(x, X0, X_E);

  float* hn_out = out + (long)T_ * B_ * 2 * H_;
  float* cn_out = hn_out + (long)2 * L_ * B_ * H_;

  for (int l = 0; l < L_; ++l) {
    (void)hipMemsetAsync(bar, 0, 256, stream);  // reset grid barrier epochs
    const unsigned short* Xin = (l == 1) ? X1 : X0;
    unsigned short* Xout = (l == 0) ? X1 : ((l == 1) ? X0 : nullptr);
    float* OutX = (l == 2) ? out : nullptr;
    lstm_layer_kernel<<<NBLK, NTHR, SMEM_BYTES, stream>>>(
        Xin, Xout, OutX,
        WihB + (long)l * 2 * 4 * H_ * DIN_,
        WhhB + (long)l * 2 * 4 * H_ * H_,
        bih + (long)l * 2 * 4 * H_,
        bhh + (long)l * 2 * 4 * H_,
        len, hbuf, hn_out, cn_out, bar, l);
  }
}
